// Attention_2808908612365
// MI455X (gfx1250) — matrix operand order repaired and checked by reading, not compiled, not run on hardware
//
#include <hip/hip_runtime.h>

// ---------------------------------------------------------------------------
// Causal multi-head attention forward for MI455X (gfx1250).
//   f16 WMMA everywhere (v_wmma_f32_16x16x32_f16), TDM (tensor_load_to_lds)
//   for the pure-f16 tile stagings, VALU fp32->f16 conversion for the rest.
// ---------------------------------------------------------------------------

#define DIMM     1024
#define HEADS    16
#define DHEAD    64
#define INNERD   1024
#define BATCH    4
#define SEQ      2048
#define MTOT     (BATCH * SEQ)      // 8192 rows total
#define QK_SCALE 0.125f             // 64^-0.5

typedef _Float16 h16;
typedef h16   v8h  __attribute__((ext_vector_type(8)));
typedef h16   v16h __attribute__((ext_vector_type(16)));
typedef float v8f  __attribute__((ext_vector_type(8)));
typedef unsigned int u32;
typedef u32   u32x4 __attribute__((ext_vector_type(4)));
typedef int   i32x4 __attribute__((ext_vector_type(4)));
typedef int   i32x8 __attribute__((ext_vector_type(8)));

#if defined(__has_builtin)
#  if __has_builtin(__builtin_amdgcn_tensor_load_to_lds) && \
      __has_builtin(__builtin_amdgcn_s_wait_tensorcnt)
#    define HAVE_TDM 1
#  endif
#endif
#ifndef HAVE_TDM
#  define HAVE_TDM 0
#endif

__device__ __forceinline__ v16h cat16(v8h lo, v8h hi) {
    v16h r;
#pragma unroll
    for (int i = 0; i < 8; ++i) { r[i] = lo[i]; r[i + 8] = hi[i]; }
    return r;
}

__device__ __forceinline__ v8f wmma_f16(v16h a, v16h b, v8f c) {
    return __builtin_amdgcn_wmma_f32_16x16x32_f16(
        false, a, false, b, (short)0, c, false, false);
}

// CDNA5 wave32 fragment layouts (ISA 7.12.2):
//  A (16x32 f16): lane L -> row (L&15); hh=L>>4; two contiguous 8-half (16B)
//                 chunks at K-offsets {8*hh, 16+8*hh}.
//  B (32x16 f16): lane L -> col (L&15); K chunks {8*hh, 16+8*hh}, as for A.
//  C/D (16x16 f32): element r at lane L -> row r+8*hh, col (L&15).

#if HAVE_TDM
// ---------------------------------------------------------------------------
// TDM: issue a 2D f16 tile DMA  global -> LDS  (ISA ch.8, D# groups 0/1).
//   pad_iv: LDS padding interval enum (2^(v+1) DWORDs), pad_am: amount enum
//   (v+1 DWORDs appended) -- reproduces the padded LDS row stride.
// One wave issues this (EXEC ignored by TDM); drain with s_wait_tensorcnt.
// ---------------------------------------------------------------------------
__device__ __forceinline__ void tdm_load_2d_f16(
        u32 lds_byte_off, const void* gptr,
        u32 tensor_d0, u32 tensor_d1, u32 stride_d0,
        u32 tile_d0, u32 tile_d1, u32 pad_iv, u32 pad_am) {
    unsigned long long ga = (unsigned long long)(size_t)gptr;
    u32x4 g0;
    g0[0] = 1u;                                            // count=1 (valid D#)
    g0[1] = lds_byte_off;                                  // lds_addr
    g0[2] = (u32)ga;                                       // global_addr[31:0]
    g0[3] = (u32)((ga >> 32) & 0x01FFFFFFu) | (2u << 30);  // addr[56:32] | type=2
    i32x8 g1;
    g1[0] = (int)((1u << 16) |                             // data_size=1 (2B)
                  (1u << 20) |                             // pad_enable
                  (pad_iv << 22) | (pad_am << 25));        // wg_mask=0
    g1[1] = (int)(tensor_d0 << 16);                        // tensor_dim0[15:0]
    g1[2] = (int)((tensor_d0 >> 16) | (tensor_d1 << 16));  // dim0 hi | dim1 lo
    g1[3] = (int)((tensor_d1 >> 16) | (tile_d0 << 16));    // dim1 hi | tile_dim0
    g1[4] = (int)(tile_d1 & 0xFFFFu);                      // tile_dim1, tile_dim2=0
    g1[5] = (int)stride_d0;                                // tensor_dim0_stride lo
    g1[6] = 0;                                             // stride hi | dim1_stride
    g1[7] = 0;
    i32x4 z4 = {0, 0, 0, 0};                               // groups 2/3 unused (2D)
#if defined(__clang_major__) && (__clang_major__ >= 23)
    i32x8 z8 = {0, 0, 0, 0, 0, 0, 0, 0};
    __builtin_amdgcn_tensor_load_to_lds(g0, g1, z4, z4, z8, 0);
#else
    __builtin_amdgcn_tensor_load_to_lds(g0, g1, z4, z4, 0);
#endif
}
#endif  // HAVE_TDM

// ---------------------------------------------------------------------------
// Kernel 1: projection GEMM  Y[m, c] = sum_k X[m,k] * W[k, colStart+c]
//   Output scattered to heads layout [B,H,N,Dh] f16, scaled.
//   256 threads = 8 waves; block tile 64(M) x 128(N); each wave a 2x2 grid of
//   16x16 WMMA tiles (fragment reuse: 4 WMMA per 4 fragment loads per K-step).
//   X and W are fp32 -> converted on the LDS store (TDM can't convert).
// ---------------------------------------------------------------------------
__global__ __launch_bounds__(256) void proj_kernel(
        const float* __restrict__ X, const float* __restrict__ W,
        int ldw, int colStart, h16* __restrict__ Y, float scale) {
    __shared__ __align__(16) h16 As[64 * 40];   // [row][k], stride 40 halves (80B)
    __shared__ __align__(16) h16 Bs[128 * 40];  // transposed: [n][k]

    const int tid  = threadIdx.x;
    const int lane = tid & 31;
    const int wave = __builtin_amdgcn_readfirstlane(tid >> 5);
    const int hh = lane >> 4, ln = lane & 15;
    const int wm = wave >> 2, wn = wave & 3;    // 2x4 waves
    const int m0 = blockIdx.x * 64;
    const int c0 = blockIdx.y * 128;

    v8f acc[2][2] = {};
    for (int k0 = 0; k0 < DIMM; k0 += 32) {
        // Stage A: 64x32 of X, fp32 -> f16, coalesced on k
#pragma unroll
        for (int i = 0; i < 8; ++i) {
            int idx = tid + i * 256;
            int r = idx >> 5, c = idx & 31;
            As[r * 40 + c] = (h16)X[(m0 + r) * DIMM + k0 + c];
        }
        // Stage B transposed: Bs[n][k] = W[k0+k][colStart+c0+n], coalesced on n
#pragma unroll
        for (int i = 0; i < 16; ++i) {
            int idx = tid + i * 256;
            int k = idx >> 7, n = idx & 127;
            Bs[n * 40 + k] = (h16)W[(k0 + k) * ldw + colStart + c0 + n];
        }
        __syncthreads();

        const int ar0 = wm * 32 + ln, ar1 = ar0 + 16;
        v16h a0 = cat16(*(const v8h*)&As[ar0 * 40 + 8 * hh],
                        *(const v8h*)&As[ar0 * 40 + 16 + 8 * hh]);
        v16h a1 = cat16(*(const v8h*)&As[ar1 * 40 + 8 * hh],
                        *(const v8h*)&As[ar1 * 40 + 16 + 8 * hh]);
        const int bn0 = wn * 32 + ln, bn1 = bn0 + 16;
        v16h b0 = cat16(*(const v8h*)&Bs[bn0 * 40 + 8 * hh],
                        *(const v8h*)&Bs[bn0 * 40 + 16 + 8 * hh]);
        v16h b1 = cat16(*(const v8h*)&Bs[bn1 * 40 + 8 * hh],
                        *(const v8h*)&Bs[bn1 * 40 + 16 + 8 * hh]);
        acc[0][0] = wmma_f16(a0, b0, acc[0][0]);
        acc[0][1] = wmma_f16(a0, b1, acc[0][1]);
        acc[1][0] = wmma_f16(a1, b0, acc[1][0]);
        acc[1][1] = wmma_f16(a1, b1, acc[1][1]);
        __syncthreads();
    }

    // Scatter C tiles into [B,H,N,Dh] f16
#pragma unroll
    for (int si = 0; si < 2; ++si)
#pragma unroll
        for (int sj = 0; sj < 2; ++sj)
#pragma unroll
            for (int r = 0; r < 8; ++r) {
                int m = m0 + wm * 32 + si * 16 + r + 8 * hh;   // b*N + n
                int c = c0 + wn * 32 + sj * 16 + ln;           // h*64 + d
                int bb = m >> 11, nn = m & (SEQ - 1);
                int hd = c >> 6,  dd = c & 63;
                Y[(((bb * HEADS + hd) * SEQ + nn) * DHEAD) + dd] =
                    (h16)(acc[si][sj][r] * scale);
            }
}

// ---------------------------------------------------------------------------
// Kernel 2: causal flash attention.
//   128 threads = 4 waves; block owns (b, h, 64-row slab); each wave a 16-row
//   Q tile. 32-key tiles: K staged by TDM (f16 tile copy, padded rows), V
//   staged transposed by VALU (TDM cannot transpose).
// ---------------------------------------------------------------------------
__global__ __launch_bounds__(128) void attn_kernel(
        const h16* __restrict__ qh, const h16* __restrict__ kh,
        const h16* __restrict__ vh, h16* __restrict__ oh) {
    __shared__ __align__(16) h16 Kt[32 * 72];      // [key][dim], stride 72 (144B)
    __shared__ __align__(16) h16 Vt[64 * 40];      // [dim][key], stride 40 (80B)
    __shared__ __align__(16) h16 Pb[4][16 * 40];   // per-wave P tile [row][key]

    const int tid  = threadIdx.x;
    const int lane = tid & 31;
    const int wave = __builtin_amdgcn_readfirstlane(tid >> 5);
    const int hh = lane >> 4, ln = lane & 15;
    const int it = blockIdx.x;
    const int h  = blockIdx.y;
    const int b  = blockIdx.z;
    const int base = (b * HEADS + h) * SEQ * DHEAD;
    const int i0 = it * 64 + wave * 16;

    // Q fragments (rows i0..i0+15, dims 0..63), pre-scaled f16.
    const h16* qrow = qh + base + (i0 + ln) * DHEAD;
    v16h aq0 = cat16(*(const v8h*)&qrow[8 * hh],      *(const v8h*)&qrow[16 + 8 * hh]);
    v16h aq1 = cat16(*(const v8h*)&qrow[32 + 8 * hh], *(const v8h*)&qrow[48 + 8 * hh]);

    v8f o0 = {}, o1 = {}, o2 = {}, o3 = {};
    float mrow[8], lrow[8];
#pragma unroll
    for (int r = 0; r < 8; ++r) { mrow[r] = -3.0e38f; lrow[r] = 0.0f; }

    const int jend = it * 64 + 32;
    for (int j0 = 0; j0 <= jend; j0 += 32) {
#if HAVE_TDM
        // K tile 32x64 f16 via TDM; LDS row = 32 DWORDs (pad_iv=4), +4 DWORDs
        // pad (pad_am=3) -> 72-half row stride matching Kt.
        if (wave == 0)
            tdm_load_2d_f16((u32)(size_t)&Kt[0], kh + base + j0 * DHEAD,
                            DHEAD, (u32)(SEQ - j0), DHEAD, DHEAD, 32, 4, 3);
#else
#pragma unroll
        for (int i = 0; i < 2; ++i) {
            int idx = tid + i * 128;
            int key = idx >> 3, c8 = (idx & 7) * 8;
            *(v8h*)&Kt[key * 72 + c8] =
                *(const v8h*)&kh[base + (j0 + key) * DHEAD + c8];
        }
#endif
        // V transposed: Vt[dim][key]
#pragma unroll
        for (int i = 0; i < 16; ++i) {
            int idx = tid + i * 128;
            int key = idx >> 6, d = idx & 63;
            Vt[d * 40 + key] = vh[base + (j0 + key) * DHEAD + d];
        }
#if HAVE_TDM
        if (wave == 0) __builtin_amdgcn_s_wait_tensorcnt(0);
#endif
        __syncthreads();

        if (j0 < i0 + 16) {                        // wave-uniform causal skip
            // S = Q K^T : two 16-key subtiles, Dh=64 split into 2 WMMAs each
            v8f s0 = {}, s1 = {};
#pragma unroll
            for (int f = 0; f < 2; ++f) {
                v16h bk0 = cat16(*(const v8h*)&Kt[ln * 72 + 32 * f + 8 * hh],
                                 *(const v8h*)&Kt[ln * 72 + 32 * f + 16 + 8 * hh]);
                v16h bk1 = cat16(*(const v8h*)&Kt[(16 + ln) * 72 + 32 * f + 8 * hh],
                                 *(const v8h*)&Kt[(16 + ln) * 72 + 32 * f + 16 + 8 * hh]);
                v16h aq = (f == 0) ? aq0 : aq1;
                s0 = wmma_f16(aq, bk0, s0);
                s1 = wmma_f16(aq, bk1, s1);
            }
            // Causal mask + online softmax (row r+8*hh = element r across lanes)
#pragma unroll
            for (int r = 0; r < 8; ++r) {
                const int irow = i0 + r + 8 * hh;
                const int jcol = j0 + ln;
                if (jcol > irow)      s0[r] = -3.0e38f;
                if (jcol + 16 > irow) s1[r] = -3.0e38f;

                float mx = fmaxf(s0[r], s1[r]);
#pragma unroll
                for (int msk = 1; msk < 16; msk <<= 1)
                    mx = fmaxf(mx, __shfl_xor(mx, msk, 32));
                float mnew = fmaxf(mrow[r], mx);
                float p0 = __expf(s0[r] - mnew);
                float p1 = __expf(s1[r] - mnew);
                float sum = p0 + p1;
#pragma unroll
                for (int msk = 1; msk < 16; msk <<= 1)
                    sum += __shfl_xor(sum, msk, 32);
                float alpha = __expf(mrow[r] - mnew);
                lrow[r] = lrow[r] * alpha + sum;
                mrow[r] = mnew;
                Pb[wave][(r + 8 * hh) * 40 + ln]      = (h16)p0;
                Pb[wave][(r + 8 * hh) * 40 + 16 + ln] = (h16)p1;
                o0[r] *= alpha; o1[r] *= alpha; o2[r] *= alpha; o3[r] *= alpha;
            }
            asm volatile("s_wait_dscnt 0" ::: "memory");   // own-wave LDS RAW

            v16h pa = cat16(*(const v8h*)&Pb[wave][ln * 40 + 8 * hh],
                            *(const v8h*)&Pb[wave][ln * 40 + 16 + 8 * hh]);
            v16h bv0 = cat16(*(const v8h*)&Vt[(0  + ln) * 40 + 8 * hh],
                             *(const v8h*)&Vt[(0  + ln) * 40 + 16 + 8 * hh]);
            v16h bv1 = cat16(*(const v8h*)&Vt[(16 + ln) * 40 + 8 * hh],
                             *(const v8h*)&Vt[(16 + ln) * 40 + 16 + 8 * hh]);
            v16h bv2 = cat16(*(const v8h*)&Vt[(32 + ln) * 40 + 8 * hh],
                             *(const v8h*)&Vt[(32 + ln) * 40 + 16 + 8 * hh]);
            v16h bv3 = cat16(*(const v8h*)&Vt[(48 + ln) * 40 + 8 * hh],
                             *(const v8h*)&Vt[(48 + ln) * 40 + 16 + 8 * hh]);
            o0 = wmma_f16(pa, bv0, o0);
            o1 = wmma_f16(pa, bv1, o1);
            o2 = wmma_f16(pa, bv2, o2);
            o3 = wmma_f16(pa, bv3, o3);
        }
        __syncthreads();
    }

    // Normalize and write attention output f16 [B, N, H*Dh]
#pragma unroll
    for (int r = 0; r < 8; ++r) {
        float inv = 1.0f / lrow[r];
        int irow = i0 + r + 8 * hh;
        int orow = (b * SEQ + irow) * INNERD + h * DHEAD;
        oh[orow + 0  + ln] = (h16)(o0[r] * inv);
        oh[orow + 16 + ln] = (h16)(o1[r] * inv);
        oh[orow + 32 + ln] = (h16)(o2[r] * inv);
        oh[orow + 48 + ln] = (h16)(o3[r] * inv);
    }
}

// ---------------------------------------------------------------------------
// Kernel 3: output projection  out = oh @ Wo + bo  (fp32 result)
//   Same 64x128 block / 2x2-per-wave tiling; A tile is already f16 so it is
//   staged by TDM (64x32 tile, padded to 40-half rows); Wo staged via VALU.
// ---------------------------------------------------------------------------
__global__ __launch_bounds__(256) void outproj_kernel(
        const h16* __restrict__ A, const float* __restrict__ Wo,
        const float* __restrict__ bo, float* __restrict__ Y) {
    __shared__ __align__(16) h16 As[64 * 40];
    __shared__ __align__(16) h16 Bs[128 * 40];

    const int tid  = threadIdx.x;
    const int lane = tid & 31;
    const int wave = __builtin_amdgcn_readfirstlane(tid >> 5);
    const int hh = lane >> 4, ln = lane & 15;
    const int wm = wave >> 2, wn = wave & 3;
    const int m0 = blockIdx.x * 64;
    const int c0 = blockIdx.y * 128;

    v8f acc[2][2] = {};
    for (int k0 = 0; k0 < INNERD; k0 += 32) {
#if HAVE_TDM
        // A tile 64 rows x 32 halves; LDS row = 16 DWORDs (pad_iv=3) + 4 DWORDs
        // pad (pad_am=3) -> 40-half row stride matching As.
        if (wave == 0)
            tdm_load_2d_f16((u32)(size_t)&As[0], A + (size_t)m0 * INNERD + k0,
                            (u32)(INNERD - k0), (u32)(MTOT - m0), INNERD, 32, 64, 3, 3);
#else
#pragma unroll
        for (int i = 0; i < 8; ++i) {
            int idx = tid + i * 256;
            int r = idx >> 5, c = idx & 31;
            As[r * 40 + c] = A[(m0 + r) * INNERD + k0 + c];
        }
#endif
#pragma unroll
        for (int i = 0; i < 16; ++i) {
            int idx = tid + i * 256;
            int k = idx >> 7, n = idx & 127;
            Bs[n * 40 + k] = (h16)Wo[(k0 + k) * DIMM + c0 + n];
        }
#if HAVE_TDM
        if (wave == 0) __builtin_amdgcn_s_wait_tensorcnt(0);
#endif
        __syncthreads();

        const int ar0 = wm * 32 + ln, ar1 = ar0 + 16;
        v16h a0 = cat16(*(const v8h*)&As[ar0 * 40 + 8 * hh],
                        *(const v8h*)&As[ar0 * 40 + 16 + 8 * hh]);
        v16h a1 = cat16(*(const v8h*)&As[ar1 * 40 + 8 * hh],
                        *(const v8h*)&As[ar1 * 40 + 16 + 8 * hh]);
        const int bn0 = wn * 32 + ln, bn1 = bn0 + 16;
        v16h b0 = cat16(*(const v8h*)&Bs[bn0 * 40 + 8 * hh],
                        *(const v8h*)&Bs[bn0 * 40 + 16 + 8 * hh]);
        v16h b1 = cat16(*(const v8h*)&Bs[bn1 * 40 + 8 * hh],
                        *(const v8h*)&Bs[bn1 * 40 + 16 + 8 * hh]);
        acc[0][0] = wmma_f16(a0, b0, acc[0][0]);
        acc[0][1] = wmma_f16(a0, b1, acc[0][1]);
        acc[1][0] = wmma_f16(a1, b0, acc[1][0]);
        acc[1][1] = wmma_f16(a1, b1, acc[1][1]);
        __syncthreads();
    }

#pragma unroll
    for (int si = 0; si < 2; ++si)
#pragma unroll
        for (int sj = 0; sj < 2; ++sj)
#pragma unroll
            for (int r = 0; r < 8; ++r) {
                int m = m0 + wm * 32 + si * 16 + r + 8 * hh;
                int c = c0 + wn * 32 + sj * 16 + ln;
                Y[m * DIMM + c] = acc[si][sj][r] + bo[c];
            }
}

// ---------------------------------------------------------------------------
// Launch: inputs (x, Wq, Wkv, Wo, bo) fp32; output fp32 [4,2048,1024].
// Workspace: q/k/v (heads layout) + attn-out as f16 = 64 MiB of d_ws.
// ---------------------------------------------------------------------------
extern "C" void kernel_launch(void* const* d_in, const int* in_sizes, int n_in,
                              void* d_out, int out_size, void* d_ws, size_t ws_size,
                              hipStream_t stream) {
    (void)in_sizes; (void)n_in; (void)out_size; (void)ws_size;
    const float* x   = (const float*)d_in[0];
    const float* Wq  = (const float*)d_in[1];
    const float* Wkv = (const float*)d_in[2];
    const float* Wo  = (const float*)d_in[3];
    const float* bo  = (const float*)d_in[4];
    float* out = (float*)d_out;

    h16* qh = (h16*)d_ws;                                   // [B,H,N,Dh]
    h16* kh = qh + (size_t)MTOT * INNERD;
    h16* vh = kh + (size_t)MTOT * INNERD;
    h16* oh = vh + (size_t)MTOT * INNERD;                   // [B,N,INNER]

    dim3 gproj(MTOT / 64, INNERD / 128);
    proj_kernel<<<gproj, 256, 0, stream>>>(x, Wq,  INNERD,     0,      qh, QK_SCALE);
    proj_kernel<<<gproj, 256, 0, stream>>>(x, Wkv, 2 * INNERD, 0,      kh, 1.0f);
    proj_kernel<<<gproj, 256, 0, stream>>>(x, Wkv, 2 * INNERD, INNERD, vh, 1.0f);

    dim3 gattn(SEQ / 64, HEADS, BATCH);
    attn_kernel<<<gattn, 128, 0, stream>>>(qh, kh, vh, oh);

    outproj_kernel<<<gproj, 256, 0, stream>>>(oh, Wo, bo, out);
}
